// GATModule_17300128268880
// MI455X (gfx1250) — matrix operand order repaired and checked by reading, not compiled, not run on hardware
//
#include <hip/hip_runtime.h>
#include <hip/hip_bf16.h>
#include <math.h>

typedef __attribute__((ext_vector_type(16))) _Float16 v16h;
typedef __attribute__((ext_vector_type(8)))  float    v8f;

#define NN   50000
#define EE   800000
#define ETOT (EE + NN)
#define IND  256
#define HEADS 4
#define CH   64
#define HC   (HEADS * CH)
#define NEG_SLOPE 0.2f

// ---------- float atomic max via ordered-int trick (lowers to native global_atomic_max/min) ----------
__device__ __forceinline__ void atomicMaxF(float* addr, float val) {
    if (val >= 0.0f) atomicMax((int*)addr, __float_as_int(val));
    else             atomicMin((unsigned int*)addr, __float_as_uint(val));
}

// ---------- 0: re-init all accumulator state (required every call for graph replay) ----------
__global__ __launch_bounds__(256) void init_kernel(float* __restrict__ agg,
                                                   float* __restrict__ m,
                                                   float* __restrict__ denom,
                                                   float* __restrict__ ewsum) {
    int t = blockIdx.x * blockDim.x + threadIdx.x;
    if (t < NN * HC) agg[t] = 0.0f;
    if (t < NN * HEADS) { m[t] = -3.0e38f; denom[t] = 0.0f; }
    if (t == 0) ewsum[0] = 0.0f;
}

// ---------- 1: per-edge weight ew = exp(-||x[src]-x[dst]||); one wave32 per edge ----------
__global__ __launch_bounds__(256) void edge_norm_kernel(const float* __restrict__ x,
                                                        const int* __restrict__ srcArr,
                                                        const int* __restrict__ dstArr,
                                                        float* __restrict__ ew,
                                                        float* __restrict__ ewsum) {
    const int wave = (blockIdx.x * blockDim.x + threadIdx.x) >> 5;
    const int lane = threadIdx.x & 31;
    const int wib  = threadIdx.x >> 5;
    __shared__ float ssum[8];
    float myw = 0.0f;
    if (wave < EE) {
        const int s = srcArr[wave];
        const int d = dstArr[wave];
        const float* __restrict__ xs = x + (size_t)s * IND + lane * 8;
        const float* __restrict__ xd = x + (size_t)d * IND + lane * 8;
        float sum = 0.0f;
#pragma unroll
        for (int j = 0; j < 8; ++j) { float t = xs[j] - xd[j]; sum = fmaf(t, t, sum); }
#pragma unroll
        for (int off = 16; off > 0; off >>= 1) sum += __shfl_xor(sum, off, 32);
        if (lane == 0) { myw = expf(-sqrtf(sum)); ew[wave] = myw; }
    }
    if (lane == 0) ssum[wib] = myw;
    __syncthreads();
    if (threadIdx.x == 0) {
        float t = 0.0f;
#pragma unroll
        for (int i = 0; i < 8; ++i) t += ssum[i];
        atomicAdd(ewsum, t);
    }
}

// ---------- 2: self-loop fill (mean ew) + per-head edge-attention coefficient ----------
__global__ __launch_bounds__(256) void coef_fill_kernel(const float* __restrict__ W_edge,
                                                        const float* __restrict__ att_edge,
                                                        const float* __restrict__ ewsum,
                                                        float* __restrict__ ew,
                                                        float* __restrict__ coef) {
    int t = blockIdx.x * blockDim.x + threadIdx.x;
    if (t < HEADS) {
        float s = 0.0f;
        for (int c = 0; c < CH; ++c) s = fmaf(W_edge[t * CH + c], att_edge[t * CH + c], s);
        coef[t] = s;
    }
    if (t < NN) ew[EE + t] = ewsum[0] * (1.0f / (float)EE);
}

// ---------- 3: h = x @ W via v_wmma_f32_16x16x32_f16; one wave per 16x16 output tile ----------
__global__ __launch_bounds__(256) void gemm_wmma_kernel(const float* __restrict__ x,
                                                        const float* __restrict__ W,
                                                        float* __restrict__ h) {
    const int wave = (blockIdx.x * blockDim.x + threadIdx.x) >> 5;
    const int lane = threadIdx.x & 31;
    const int tileRow = wave >> 4;      // 3125 row tiles (50000/16)
    const int tileCol = wave & 15;      // 16 col tiles (256/16)
    if (tileRow >= NN / 16) return;     // wave-uniform; EXEC stays all-1s for WMMA
    const int half = lane >> 4;
    const int mrow = lane & 15;
    const int row  = tileRow * 16 + mrow;
    const int col  = tileCol * 16 + mrow;
    const float* __restrict__ arow = x + (size_t)row * IND;
    v8f acc = {0.f, 0.f, 0.f, 0.f, 0.f, 0.f, 0.f, 0.f};
    for (int k0 = 0; k0 < IND; k0 += 32) {
        v16h a, b;
        const int ka = k0 + 8 * half;           // A: 16-bit 16x32 layout (ISA 7.12.2)
#pragma unroll
        for (int j = 0; j < 8; ++j) a[j]     = (_Float16)arow[ka + j];
#pragma unroll
        for (int j = 0; j < 8; ++j) a[8 + j] = (_Float16)arow[ka + 16 + j];
        const int kb = k0 + 8 * half;           // B: 32x16, b[i] = W[kb+(i&7)+16*(i>>3)][col]
#pragma unroll
        for (int i = 0; i < 16; ++i) b[i] = (_Float16)W[(size_t)(kb + (i & 7) + ((i >> 3) << 4)) * HC + col];
        acc = __builtin_amdgcn_wmma_f32_16x16x32_f16(false, a, false, b, (short)0, acc,
                                                     false, false);
    }
#pragma unroll
    for (int r = 0; r < 8; ++r)                 // D: acc[r] = D[r + 8*half][mrow]
        h[(size_t)(tileRow * 16 + r + 8 * half) * HC + col] = acc[r];
}

// ---------- 4: per-node attention logits a_src/a_dst ----------
__global__ __launch_bounds__(256) void node_att_kernel(const float* __restrict__ h,
                                                       const float* __restrict__ att_src,
                                                       const float* __restrict__ att_dst,
                                                       float* __restrict__ a_src,
                                                       float* __restrict__ a_dst) {
    int t = blockIdx.x * blockDim.x + threadIdx.x;
    if (t >= NN * HEADS) return;
    int n = t >> 2, hh = t & 3;
    const float* __restrict__ hp = h + (size_t)n * HC + hh * CH;
    const float* __restrict__ as = att_src + hh * CH;
    const float* __restrict__ ad = att_dst + hh * CH;
    float s1 = 0.0f, s2 = 0.0f;
#pragma unroll 8
    for (int c = 0; c < CH; ++c) { s1 = fmaf(hp[c], as[c], s1); s2 = fmaf(hp[c], ad[c], s2); }
    a_src[t] = s1;
    a_dst[t] = s2;
}

// ---------- 5: alpha = leaky_relu(...); segment max into m[dst] ----------
__global__ __launch_bounds__(256) void alpha_max_kernel(const int* __restrict__ srcArr,
                                                        const int* __restrict__ dstArr,
                                                        const float* __restrict__ ew,
                                                        const float* __restrict__ a_src,
                                                        const float* __restrict__ a_dst,
                                                        const float* __restrict__ coef,
                                                        float* __restrict__ alpha,
                                                        float* __restrict__ m) {
    int t = blockIdx.x * blockDim.x + threadIdx.x;
    if (t >= ETOT * HEADS) return;
    int e = t >> 2, hh = t & 3;
    int s = (e < EE) ? srcArr[e] : (e - EE);
    int d = (e < EE) ? dstArr[e] : (e - EE);
    float v = a_src[s * HEADS + hh] + a_dst[d * HEADS + hh] + ew[e] * coef[hh];
    v = (v > 0.0f) ? v : NEG_SLOPE * v;
    alpha[t] = v;
    atomicMaxF(&m[d * HEADS + hh], v);
}

// ---------- 6: ex = exp(alpha - m[dst]); segment sum into denom ----------
__global__ __launch_bounds__(256) void exp_sum_kernel(const int* __restrict__ dstArr,
                                                      const float* __restrict__ m,
                                                      float* __restrict__ alpha,
                                                      float* __restrict__ denom) {
    int t = blockIdx.x * blockDim.x + threadIdx.x;
    if (t >= ETOT * HEADS) return;
    int e = t >> 2, hh = t & 3;
    int d = (e < EE) ? dstArr[e] : (e - EE);
    float ex = expf(alpha[t] - m[d * HEADS + hh]);
    alpha[t] = ex;
    atomicAdd(&denom[d * HEADS + hh], ex);
}

// ---------- 7: weighted scatter-aggregate; one wave32 per edge, 8 channels/lane ----------
__global__ __launch_bounds__(256) void aggregate_kernel(const int* __restrict__ srcArr,
                                                        const int* __restrict__ dstArr,
                                                        const float* __restrict__ h,
                                                        const float* __restrict__ alpha,
                                                        const float* __restrict__ denom,
                                                        float* __restrict__ agg) {
    const int wave = (blockIdx.x * blockDim.x + threadIdx.x) >> 5;
    const int lane = threadIdx.x & 31;
    if (wave >= ETOT) return;
    int s = (wave < EE) ? srcArr[wave] : (wave - EE);
    int d = (wave < EE) ? dstArr[wave] : (wave - EE);
    int hh = lane >> 3;   // lane's 8 channels fall inside one head (64ch / 8 lanes)
    float w = alpha[wave * HEADS + hh] / (denom[d * HEADS + hh] + 1e-16f);
    const float* __restrict__ hp = h + (size_t)s * HC + lane * 8;
    float* __restrict__ ap = agg + (size_t)d * HC + lane * 8;
#pragma unroll
    for (int j = 0; j < 8; ++j) atomicAdd(ap + j, hp[j] * w);
}

// ---------- 8: head mean + bias + relu ----------
__global__ __launch_bounds__(256) void finalize_kernel(const float* __restrict__ agg,
                                                       const float* __restrict__ bias,
                                                       float* __restrict__ out) {
    int t = blockIdx.x * blockDim.x + threadIdx.x;
    if (t >= NN * CH) return;
    int n = t >> 6, c = t & 63;
    const float* __restrict__ a = agg + (size_t)n * HC;
    float v = 0.25f * (a[c] + a[CH + c] + a[2 * CH + c] + a[3 * CH + c]) + bias[c];
    out[t] = fmaxf(v, 0.0f);
}

extern "C" void kernel_launch(void* const* d_in, const int* in_sizes, int n_in,
                              void* d_out, int out_size, void* d_ws, size_t ws_size,
                              hipStream_t stream) {
    const float* x        = (const float*)d_in[0];
    const int*   ei       = (const int*)d_in[1];   // [2,E]: src = ei[0:E], dst = ei[E:2E]
    const float* W        = (const float*)d_in[2];
    const float* att_src  = (const float*)d_in[3];
    const float* att_dst  = (const float*)d_in[4];
    const float* W_edge   = (const float*)d_in[5];
    const float* att_edge = (const float*)d_in[6];
    const float* bias     = (const float*)d_in[7];
    float*       out      = (float*)d_out;

    const int* srcArr = ei;
    const int* dstArr = ei + EE;

    // workspace layout (floats)
    float* ws    = (float*)d_ws;
    float* h     = ws;                                   // N*HC   = 12.8M
    float* agg   = h     + (size_t)NN * HC;              // N*HC   = 12.8M
    float* ew    = agg   + (size_t)NN * HC;              // ETOT   = 0.85M
    float* alpha = ew    + (size_t)ETOT;                 // ETOT*H = 3.4M
    float* a_src = alpha + (size_t)ETOT * HEADS;         // N*H
    float* a_dst = a_src + (size_t)NN * HEADS;           // N*H
    float* m     = a_dst + (size_t)NN * HEADS;           // N*H
    float* denom = m     + (size_t)NN * HEADS;           // N*H
    float* coef  = denom + (size_t)NN * HEADS;           // H
    float* ewsum = coef  + HEADS;                        // 1

    const int T = 256;
    init_kernel     <<<(NN * HC + T - 1) / T,      T, 0, stream>>>(agg, m, denom, ewsum);
    edge_norm_kernel<<<EE / 8,                     T, 0, stream>>>(x, srcArr, dstArr, ew, ewsum);
    coef_fill_kernel<<<(NN + T - 1) / T,           T, 0, stream>>>(W_edge, att_edge, ewsum, ew, coef);
    gemm_wmma_kernel<<<(NN / 16) * (HC / 16) / 8,  T, 0, stream>>>(x, W, h);
    node_att_kernel <<<(NN * HEADS + T - 1) / T,   T, 0, stream>>>(h, att_src, att_dst, a_src, a_dst);
    alpha_max_kernel<<<(ETOT * HEADS + T - 1) / T, T, 0, stream>>>(srcArr, dstArr, ew, a_src, a_dst,
                                                                   coef, alpha, m);
    exp_sum_kernel  <<<(ETOT * HEADS + T - 1) / T, T, 0, stream>>>(dstArr, m, alpha, denom);
    aggregate_kernel<<<ETOT / 8,                   T, 0, stream>>>(srcArr, dstArr, h, alpha, denom, agg);
    finalize_kernel <<<(NN * CH + T - 1) / T,      T, 0, stream>>>(agg, bias, out);
}
